// BasicDMPNN_326417514983
// MI455X (gfx1250) — matrix operand order repaired and checked by reading, not compiled, not run on hardware
//
#include <hip/hip_runtime.h>
#include <hip/hip_bf16.h>

typedef __attribute__((ext_vector_type(16))) _Float16 v16h;
typedef __attribute__((ext_vector_type(8)))  float    v8f;
typedef __attribute__((ext_vector_type(4)))  float    v4f;

#define N_NODES 50000
#define N_EDGES 800000
#define NUM_MOL 2000
#define ETILES  (N_EDGES / 16)   // 50000, 16 edges per WMMA M-tile

// fragment geometry
#define NT 8          // 128 output cols / 16
#define KT_AB 6       // 192 / 32
#define KT_U1 10      // 320 / 32
#define KT_U2 4       // 128 / 32

__device__ __forceinline__ v8f wmma_f16(v16h a, v16h b, v8f c) {
  return __builtin_amdgcn_wmma_f32_16x16x32_f16(false, a, false, b, (short)0, c, false, false);
}

// Build a 16-bit A-fragment (ISA 7.12.2 16x32 layout) from two contiguous
// 8-float segments at k0 = kt*32 + 8*kh and k0+16 (each lane: row m = lane&15).
__device__ __forceinline__ v16h make_afrag(const float* seg0, const float* seg1) {
  v4f s0 = *(const v4f*)(seg0);
  v4f s1 = *(const v4f*)(seg0 + 4);
  v4f s2 = *(const v4f*)(seg1);
  v4f s3 = *(const v4f*)(seg1 + 4);
  v16h a;
#pragma unroll
  for (int i = 0; i < 4; ++i) {
    a[i]      = (_Float16)s0[i];
    a[4 + i]  = (_Float16)s1[i];
    a[8 + i]  = (_Float16)s2[i];
    a[12 + i] = (_Float16)s3[i];
  }
  return a;
}

// ---------------------------------------------------------------------------
// Pack weights (f32 row-major [K,128]) into B-fragment layout f16:
// addr = ((kt*NT + nt)*32 + lane)*16 + h ; k = kt*32 + 8*(lane>>4) + (h<8 ? h : h+8), n = nt*16 + (lane&15)
// ---------------------------------------------------------------------------
__global__ void prep_weights_kernel(const float* __restrict__ Wi,
                                    const float* __restrict__ Wu1,
                                    const float* __restrict__ Wu2,
                                    _Float16* __restrict__ wif,
                                    _Float16* __restrict__ wu1f,
                                    _Float16* __restrict__ wu2f) {
  int idx = blockIdx.x * blockDim.x + threadIdx.x;
  const int n0 = 192 * 128, n1 = 320 * 128, n2 = 128 * 128;
  const float* W; _Float16* D; int li;
  if (idx < n0)            { W = Wi;  D = wif;  li = idx; }
  else if (idx < n0 + n1)  { W = Wu1; D = wu1f; li = idx - n0; }
  else if (idx < n0+n1+n2) { W = Wu2; D = wu2f; li = idx - n0 - n1; }
  else return;
  int h = li & 15, lane = (li >> 4) & 31, t2 = li >> 9;
  int nt = t2 & 7, kt = t2 >> 3;
  int k = kt * 32 + 8 * (lane >> 4) + (h < 8 ? h : h + 8);
  int n = nt * 16 + (lane & 15);
  D[li] = (_Float16)W[k * 128 + n];
}

__global__ void zero_f32_kernel(float* __restrict__ p, int n) {
  int i = blockIdx.x * blockDim.x + threadIdx.x;
  if (i < n) p[i] = 0.0f;
}

// Per-lane segment pointers for the ab = [atom_table[x[src]] | bond_table[ea]]
// virtual row: segment of 8 floats starting at k0 (never straddles the 128 split).
__device__ __forceinline__ const float* ab_seg(const float* arow, const float* brow, int k0) {
  return (k0 < 128) ? (arow + k0) : (brow + (k0 - 128));
}

// ---------------------------------------------------------------------------
// Initial pass: msg0 = relu(ab @ Wi + bi); agg1 = scatter_add(msg0, dst)
// A-fragments built inline from the (L2/WGP$-resident) embedding tables.
// ---------------------------------------------------------------------------
__global__ __launch_bounds__(256) void init_gemm_kernel(
    const int* __restrict__ x, const int* __restrict__ eattr,
    const int* __restrict__ eindex,
    const float* __restrict__ atomT, const float* __restrict__ bondT,
    const _Float16* __restrict__ wif, const float* __restrict__ bi,
    float* __restrict__ agg) {
  __shared__ __attribute__((aligned(32))) _Float16 sWi[KT_AB * NT * 512];
  {
    const uint4* s = (const uint4*)wif; uint4* d = (uint4*)sWi;
    for (int i = threadIdx.x; i < (KT_AB * NT * 512) / 8; i += 256) d[i] = s[i];
  }
  __syncthreads();
  int wave = threadIdx.x >> 5, lane = threadIdx.x & 31;
  int tile = blockIdx.x * 8 + wave;
  if (tile >= ETILES) return;
  int ln = lane & 15, kh = lane >> 4;
  int e = tile * 16 + ln;                       // A-row owned by this lane
  const float* arow = atomT + (size_t)x[eindex[e]] * 128;
  const float* brow = bondT + (size_t)eattr[e] * 64;

  v8f acc[NT];
#pragma unroll
  for (int nt = 0; nt < NT; ++nt) {
    float b = bi[nt * 16 + ln];
#pragma unroll
    for (int r = 0; r < 8; ++r) acc[nt][r] = b;
  }
#pragma unroll
  for (int kt = 0; kt < KT_AB; ++kt) {
    int k0 = kt * 32 + 8 * kh;
    v16h a = make_afrag(ab_seg(arow, brow, k0), ab_seg(arow, brow, k0 + 16));
#pragma unroll
    for (int nt = 0; nt < NT; ++nt) {
      v16h b = *(const v16h*)(sWi + (kt * NT + nt) * 512 + lane * 16);
      acc[nt] = wmma_f16(a, b, acc[nt]);
    }
  }
  const int* dstp = eindex + N_EDGES;
  int ebase = tile * 16;
#pragma unroll
  for (int r = 0; r < 8; ++r) {
    float* orow = agg + (size_t)dstp[ebase + r + 8 * kh] * 128;
#pragma unroll
    for (int nt = 0; nt < NT; ++nt)
      unsafeAtomicAdd(orow + nt * 16 + ln, fmaxf(acc[nt][r], 0.0f));
  }
}

// ---------------------------------------------------------------------------
// Fused per-pass kernel:
//   h   = relu([ab | aggIn[src]] @ Wu1 + bu1)   (K=320; A built inline from
//                                                L2-resident tables + agg)
//   msg = relu(h @ Wu2 + bu2)                   (K=128, via LDS re-layout)
//   aggOut[dst] += msg                          (segment_sum via f32 atomics)
// ---------------------------------------------------------------------------
__global__ __launch_bounds__(256) void pass_gemm_kernel(
    const int* __restrict__ x, const int* __restrict__ eattr,
    const int* __restrict__ eindex,
    const float* __restrict__ atomT, const float* __restrict__ bondT,
    const float* __restrict__ aggIn,
    const _Float16* __restrict__ wu1f, const _Float16* __restrict__ wu2f,
    const float* __restrict__ bu1, const float* __restrict__ bu2,
    float* __restrict__ aggOut) {
  __shared__ __attribute__((aligned(32))) _Float16 sW1[KT_U1 * NT * 512]; // 80KB
  __shared__ __attribute__((aligned(32))) _Float16 sW2[KT_U2 * NT * 512]; // 32KB
  __shared__ __attribute__((aligned(32))) _Float16 sH[8 * 16 * 128];      // 32KB
  {
    const uint4* s1 = (const uint4*)wu1f; uint4* d1 = (uint4*)sW1;
    for (int i = threadIdx.x; i < (KT_U1 * NT * 512) / 8; i += 256) d1[i] = s1[i];
    const uint4* s2 = (const uint4*)wu2f; uint4* d2 = (uint4*)sW2;
    for (int i = threadIdx.x; i < (KT_U2 * NT * 512) / 8; i += 256) d2[i] = s2[i];
  }
  __syncthreads();
  int wave = threadIdx.x >> 5, lane = threadIdx.x & 31;
  int tile = blockIdx.x * 8 + wave;
  if (tile >= ETILES) return;
  int ln = lane & 15, kh = lane >> 4;
  int e = tile * 16 + ln;                       // A-row owned by this lane
  int src = eindex[e];
  const float* arow = atomT + (size_t)x[src] * 128;
  const float* brow = bondT + (size_t)eattr[e] * 64;
  const float* grow = aggIn + (size_t)src * 128;

  // ---- GEMM1: K = 320 = 6 ab-tiles + 4 gathered-agg tiles ----
  v8f acc[NT];
#pragma unroll
  for (int nt = 0; nt < NT; ++nt) {
    float b = bu1[nt * 16 + ln];
#pragma unroll
    for (int r = 0; r < 8; ++r) acc[nt][r] = b;
  }
#pragma unroll
  for (int kt = 0; kt < KT_U1; ++kt) {
    v16h a;
    if (kt < KT_AB) {
      int k0 = kt * 32 + 8 * kh;
      a = make_afrag(ab_seg(arow, brow, k0), ab_seg(arow, brow, k0 + 16));
    } else {
      int k0 = (kt - KT_AB) * 32 + 8 * kh;
      a = make_afrag(grow + k0, grow + k0 + 16);
    }
#pragma unroll
    for (int nt = 0; nt < NT; ++nt) {
      v16h b = *(const v16h*)(sW1 + (kt * NT + nt) * 512 + lane * 16);
      acc[nt] = wmma_f16(a, b, acc[nt]);
    }
  }

  // ---- ReLU + wave-private LDS re-layout (D layout -> row-major 16x128) ----
  _Float16* hrow = sH + wave * 2048;
#pragma unroll
  for (int nt = 0; nt < NT; ++nt)
#pragma unroll
    for (int r = 0; r < 8; ++r)
      hrow[(r + 8 * kh) * 128 + nt * 16 + ln] = (_Float16)fmaxf(acc[nt][r], 0.0f);

  // ---- GEMM2: K = 128, A fragments re-read from LDS (packed K-pairs) ----
  v8f acc2[NT];
#pragma unroll
  for (int nt = 0; nt < NT; ++nt) {
    float b = bu2[nt * 16 + ln];
#pragma unroll
    for (int r = 0; r < 8; ++r) acc2[nt][r] = b;
  }
#pragma unroll
  for (int kt = 0; kt < KT_U2; ++kt) {
    v16h a;
    unsigned int* ap = (unsigned int*)&a;
#pragma unroll
    for (int p = 0; p < 8; ++p) {
      int k = kt * 32 + (p < 4 ? 2 * p : 2 * p + 8) + 8 * kh;
      ap[p] = *(const unsigned int*)(hrow + ln * 128 + k);
    }
#pragma unroll
    for (int nt = 0; nt < NT; ++nt) {
      v16h b = *(const v16h*)(sW2 + (kt * NT + nt) * 512 + lane * 16);
      acc2[nt] = wmma_f16(a, b, acc2[nt]);
    }
  }

  // ---- ReLU + segment_sum scatter (atomics land in L2-resident aggOut) ----
  const int* dstp = eindex + N_EDGES;
  int ebase = tile * 16;
#pragma unroll
  for (int r = 0; r < 8; ++r) {
    float* orow = aggOut + (size_t)dstp[ebase + r + 8 * kh] * 128;
#pragma unroll
    for (int nt = 0; nt < NT; ++nt)
      unsafeAtomicAdd(orow + nt * 16 + ln, fmaxf(acc2[nt][r], 0.0f));
  }
}

// ---------------------------------------------------------------------------
// mol_state[batch[i]] += node_state[i]
// ---------------------------------------------------------------------------
__global__ void mol_reduce_kernel(const float* __restrict__ nodeS,
                                  const int* __restrict__ batch,
                                  float* __restrict__ mol) {
  int idx = blockIdx.x * blockDim.x + threadIdx.x;
  if (idx >= N_NODES * 128) return;
  int i = idx >> 7, c = idx & 127;
  unsafeAtomicAdd(mol + (size_t)batch[i] * 128 + c, nodeS[idx]);
}

// ---------------------------------------------------------------------------
// out = relu(mol @ Wr1 + br1) @ Wr2 + br2   (tiny: plain VALU)
// ---------------------------------------------------------------------------
__global__ void readout_kernel(const float* __restrict__ mol,
                               const float* __restrict__ Wr1,
                               const float* __restrict__ br1,
                               const float* __restrict__ Wr2,
                               const float* __restrict__ br2,
                               float* __restrict__ out) {
  __shared__ float mr[128];
  __shared__ float red[256];
  int m = blockIdx.x, t = threadIdx.x;
  if (t < 128) mr[t] = mol[(size_t)m * 128 + t];
  __syncthreads();
  float p = 0.0f;
#pragma unroll
  for (int jj = 0; jj < 2; ++jj) {
    int j = t + jj * 256;
    float s = br1[j];
    for (int k = 0; k < 128; ++k) s += mr[k] * Wr1[k * 512 + j];
    p += fmaxf(s, 0.0f) * Wr2[j];
  }
  red[t] = p;
  __syncthreads();
  for (int off = 128; off > 0; off >>= 1) {
    if (t < off) red[t] += red[t + off];
    __syncthreads();
  }
  if (t == 0) out[m] = red[0] + br2[0];
}

// ---------------------------------------------------------------------------
extern "C" void kernel_launch(void* const* d_in, const int* in_sizes, int n_in,
                              void* d_out, int out_size, void* d_ws, size_t ws_size,
                              hipStream_t stream) {
  const int*   x      = (const int*)d_in[0];
  const int*   eattr  = (const int*)d_in[1];
  const int*   eindex = (const int*)d_in[2];
  const int*   batch  = (const int*)d_in[3];
  const float* atomT  = (const float*)d_in[4];
  const float* bondT  = (const float*)d_in[5];
  const float* Wi     = (const float*)d_in[6];
  const float* bi     = (const float*)d_in[7];
  const float* Wu1    = (const float*)d_in[8];
  const float* bu1    = (const float*)d_in[9];
  const float* Wu2    = (const float*)d_in[10];
  const float* bu2    = (const float*)d_in[11];
  const float* Wr1    = (const float*)d_in[12];
  const float* br1    = (const float*)d_in[13];
  const float* Wr2    = (const float*)d_in[14];
  const float* br2    = (const float*)d_in[15];
  float* out = (float*)d_out;

  char* ws = (char*)d_ws;
  size_t off = 0;
  float*    aggA  = (float*)(ws + off);    off += (size_t)N_NODES * 128 * 4; // 25.6MB
  float*    aggB  = (float*)(ws + off);    off += (size_t)N_NODES * 128 * 4; // 25.6MB
  _Float16* wif   = (_Float16*)(ws + off); off += 192 * 128 * 2;
  _Float16* wu1f  = (_Float16*)(ws + off); off += 320 * 128 * 2;
  _Float16* wu2f  = (_Float16*)(ws + off); off += 128 * 128 * 2;
  float*    mol   = (float*)(ws + off);    off += (size_t)NUM_MOL * 128 * 4;

  const int gemmBlocks = ETILES / 8;                 // 6250
  const int aggElems   = N_NODES * 128;              // 6.4M
  const int aggBlocks  = (aggElems + 255) / 256;

  prep_weights_kernel<<<320, 256, 0, stream>>>(Wi, Wu1, Wu2, wif, wu1f, wu2f);

  zero_f32_kernel<<<aggBlocks, 256, 0, stream>>>(aggA, aggElems);
  init_gemm_kernel<<<gemmBlocks, 256, 0, stream>>>(x, eattr, eindex, atomT, bondT,
                                                   wif, bi, aggA);

  float* aggIn = aggA;
  float* aggOut = aggB;
  for (int p = 0; p < 4; ++p) {
    zero_f32_kernel<<<aggBlocks, 256, 0, stream>>>(aggOut, aggElems);
    pass_gemm_kernel<<<gemmBlocks, 256, 0, stream>>>(x, eattr, eindex, atomT, bondT,
                                                     aggIn, wu1f, wu2f,
                                                     bu1, bu2, aggOut);
    float* t = aggIn; aggIn = aggOut; aggOut = t;
  }
  // aggIn now holds node_state (the 5th segment_sum)

  zero_f32_kernel<<<(NUM_MOL * 128 + 255) / 256, 256, 0, stream>>>(mol, NUM_MOL * 128);
  mol_reduce_kernel<<<aggBlocks, 256, 0, stream>>>(aggIn, batch, mol);
  readout_kernel<<<NUM_MOL, 256, 0, stream>>>(mol, Wr1, br1, Wr2, br2, out);
}
